// LeafPositionalEncoding_21036749815919
// MI455X (gfx1250) — matrix operand order repaired and checked by reading, not compiled, not run on hardware
//
#include <hip/hip_runtime.h>
#include <math.h>

typedef __bf16 bf16;
typedef __attribute__((ext_vector_type(8)))  bf16  v8bf;
typedef __attribute__((ext_vector_type(16))) bf16  v16bf;
typedef __attribute__((ext_vector_type(8)))  float v8f;

#define HIDDEN 1024
#define HEADS 16
#define DK 64
#define NBUCK 32

// ---------------------------------------------------------------------------
// CDNA5 async global->LDS copies (ASYNCcnt).  The clang builtin takes
// GCC-vector int4 pointers: (AS1 v4i* gsrc, AS3 v4i* ldst, imm off, imm cpol);
// INST_OFFSET is added to BOTH addresses (ISA 08 §4.4).
// ---------------------------------------------------------------------------
#if defined(__gfx1250__) && __has_builtin(__builtin_amdgcn_global_load_async_to_lds_b128)
#define HAVE_ASYNC_LDS 1
#else
#define HAVE_ASYNC_LDS 0
#endif

#if HAVE_ASYNC_LDS
typedef int v4i __attribute__((vector_size(16)));
#define AS1 __attribute__((address_space(1)))
#define AS3 __attribute__((address_space(3)))

__device__ __forceinline__ void async_copy32(const void* g, void* l) {
  AS1 v4i* gp = (AS1 v4i*)g;
  AS3 v4i* lp = (AS3 v4i*)l;
  __builtin_amdgcn_global_load_async_to_lds_b128(gp, lp, 0, 0);
  __builtin_amdgcn_global_load_async_to_lds_b128(gp, lp, 16, 0);
}
__device__ __forceinline__ void async_copy64(const void* g, void* l) {
  AS1 v4i* gp = (AS1 v4i*)g;
  AS3 v4i* lp = (AS3 v4i*)l;
  __builtin_amdgcn_global_load_async_to_lds_b128(gp, lp, 0, 0);
  __builtin_amdgcn_global_load_async_to_lds_b128(gp, lp, 16, 0);
  __builtin_amdgcn_global_load_async_to_lds_b128(gp, lp, 32, 0);
  __builtin_amdgcn_global_load_async_to_lds_b128(gp, lp, 48, 0);
}
__device__ __forceinline__ void wait_async0() {
#if __has_builtin(__builtin_amdgcn_s_wait_asynccnt)
  __builtin_amdgcn_s_wait_asynccnt(0);
#else
  asm volatile("s_wait_asynccnt 0x0" ::: "memory");
#endif
}
#endif

// ---------------------------------------------------------------------------
// Fragment loaders (wave32); the k order below is this code's choice, used alike for A and B.
// A (16x32 bf16, MxK): lane&15 = M row; lanes 0-15 hold K[0..16),
//                      lanes 16-31 hold K[16..32)  (contiguous in K).
// B (32x16 bf16, KxN): lane&15 = N col; lanes 0-15 hold K[0..16),
//                      lanes 16-31 hold K[16..32)  (contiguous in K).
// So B loads directly from a row-major [N][K] matrix.
// ---------------------------------------------------------------------------
__device__ __forceinline__ v16bf frag_a(const bf16* base, int ld, int mbase,
                                        int kbase, int lane) {
  const bf16* p = base + (mbase + (lane & 15)) * ld + kbase + 16 * (lane >> 4);
  v8bf lo = *(const v8bf*)(p);
  v8bf hi = *(const v8bf*)(p + 8);
  v16bf r;
#pragma unroll
  for (int t = 0; t < 8; ++t) { r[t] = lo[t]; r[t + 8] = hi[t]; }
  return r;
}

__device__ __forceinline__ v16bf frag_b(const bf16* base, int ld, int nbase,
                                        int kbase, int lane) {
  const bf16* p = base + (nbase + (lane & 15)) * ld + kbase + 16 * (lane >> 4);
  return *(const v16bf*)(p);
}

__device__ __forceinline__ v8f wmma_bf16(v16bf a, v16bf b, v8f c) {
  return __builtin_amdgcn_wmma_f32_16x16x32_bf16(false, a, false, b,
                                                 (short)0, c, false, false);
}

// ---------------------------------------------------------------------------
// Kernel 1: T5 relative-position bucket LUT.  idx = rel + 2047, rel in
// [-2047, 2047].  Only place logf is evaluated (4K times, not 64M).
// ---------------------------------------------------------------------------
__global__ void bucket_lut_kernel(unsigned char* tab) {
  int idx = blockIdx.x * blockDim.x + threadIdx.x;   // 0..4095
  int rel = idx - 2047;
  int n = -rel;
  int ret = (n < 0) ? (NBUCK / 2) : 0;               // 16
  if (n < 0) n = -n;
  int bucket;
  if (n < 8) {
    bucket = n;
  } else {
    // scale = (16-8)/ln(128/8) = 8/ln(16)
    float v = logf((float)n * 0.125f) * 2.8853900817779268f;
    int vl = 8 + (int)v;                             // trunc, matches astype
    bucket = vl < 15 ? vl : 15;
  }
  tab[idx] = (unsigned char)(ret + bucket);
}

// ---------------------------------------------------------------------------
// Kernel 2: convert Wq / Wk f32 -> bf16 (1M elements each).
// ---------------------------------------------------------------------------
__global__ void cvt_w_kernel(const float* __restrict__ wq,
                             const float* __restrict__ wk,
                             bf16* __restrict__ oq, bf16* __restrict__ ok) {
  int i = blockIdx.x * blockDim.x + threadIdx.x;     // 0..1048575
  oq[i] = (bf16)wq[i];
  ok[i] = (bf16)wk[i];
}

// ---------------------------------------------------------------------------
// Kernel 3: gather p_emb[leaf] + LayerNorm, emit bf16 xn[4096][1024].
// One wave32 per row; 32 floats per lane; shfl reductions.
// ---------------------------------------------------------------------------
__global__ __launch_bounds__(256) void ln_kernel(
    const int* __restrict__ leaf, const float* __restrict__ pemb,
    const float* __restrict__ g, const float* __restrict__ bta,
    bf16* __restrict__ xn) {
  int lane = threadIdx.x & 31;
  int row = blockIdx.x * 8 + (threadIdx.x >> 5);     // 4096 rows
  int idx = leaf[row];
  const float* src = pemb + (size_t)idx * HIDDEN;
  float x[32];
  float s = 0.f;
#pragma unroll
  for (int c = 0; c < 32; ++c) { x[c] = src[c * 32 + lane]; s += x[c]; }
#pragma unroll
  for (int off = 16; off > 0; off >>= 1) s += __shfl_xor(s, off, 32);
  float mu = s * (1.0f / HIDDEN);
  float v = 0.f;
#pragma unroll
  for (int c = 0; c < 32; ++c) { float d = x[c] - mu; v += d * d; }
#pragma unroll
  for (int off = 16; off > 0; off >>= 1) v += __shfl_xor(v, off, 32);
  float inv = rsqrtf(v * (1.0f / HIDDEN) + 1e-5f);
  bf16* dst = xn + (size_t)row * HIDDEN;
#pragma unroll
  for (int c = 0; c < 32; ++c) {
    int e = c * 32 + lane;
    dst[e] = (bf16)((x[c] - mu) * inv * g[e] + bta[e]);
  }
}

// ---------------------------------------------------------------------------
// Kernel 4: fused Q / K projection.  out = xn @ W^T + bias (Q also *1/8).
// M=4096, N=1024, K=1024.  grid = (8, 32, 2); z picks Wq vs Wk.
// Block tile 128x128, 8 waves, wave w owns 16-column strip, 8 m-tiles.
// Async double-buffered LDS staging (ASYNCcnt) when available.
// Output scattered into per-head layout q[bh][i][d] (bf16).
// ---------------------------------------------------------------------------
__global__ __launch_bounds__(256) void proj_kernel(
    const bf16* __restrict__ xn, const bf16* __restrict__ wqb,
    const bf16* __restrict__ wkb, const float* __restrict__ bq,
    const float* __restrict__ bk, bf16* __restrict__ qout,
    bf16* __restrict__ kout) {
  const int which = blockIdx.z;
  const bf16* W = which ? wkb : wqb;
  const float* bias = which ? bk : bq;
  bf16* out = which ? kout : qout;
  const float scale = which ? 1.0f : 0.125f;         // 1/sqrt(64) for Q

#if HAVE_ASYNC_LDS
  __shared__ __align__(32) bf16 As[2][128][40];      // ping-pong, +8 pad
  __shared__ __align__(32) bf16 Bs[2][128][40];
#else
  __shared__ __align__(32) bf16 As[1][128][40];
  __shared__ __align__(32) bf16 Bs[1][128][40];
#endif

  const int tid = threadIdx.x, lane = tid & 31, wv = tid >> 5;
  const int m0 = blockIdx.y * 128, n0 = blockIdx.x * 128;

  v8f acc[8];
#pragma unroll
  for (int t = 0; t < 8; ++t) acc[t] = (v8f)0.0f;

  const int lr = tid >> 1;               // 128 rows, 2 threads per row
  const int lc = (tid & 1) * 16;         // 16 halfs (32B) each
  const bf16* ga = xn + (size_t)(m0 + lr) * HIDDEN + lc;
  const bf16* gb = W + (size_t)(n0 + lr) * HIDDEN + lc;

#if HAVE_ASYNC_LDS
  async_copy32(ga, &As[0][lr][lc]);
  async_copy32(gb, &Bs[0][lr][lc]);
  wait_async0();
  __syncthreads();
  for (int k0 = 0; k0 < HIDDEN; k0 += 32) {
    const int buf = (k0 >> 5) & 1;
    if (k0 + 32 < HIDDEN) {              // prefetch next K-tile into other buf
      async_copy32(ga + k0 + 32, &As[buf ^ 1][lr][lc]);
      async_copy32(gb + k0 + 32, &Bs[buf ^ 1][lr][lc]);
    }
    v16bf bfrg = frag_b(&Bs[buf][0][0], 40, wv * 16, 0, lane);
#pragma unroll
    for (int t = 0; t < 8; ++t) {
      v16bf af = frag_a(&As[buf][0][0], 40, t * 16, 0, lane);
      acc[t] = wmma_bf16(af, bfrg, acc[t]);
    }
    wait_async0();                        // next tile landed in LDS
    __syncthreads();                      // everyone done reading this buf
  }
#else
  for (int k0 = 0; k0 < HIDDEN; k0 += 32) {
    *(v16bf*)&As[0][lr][lc] = *(const v16bf*)(ga + k0);
    *(v16bf*)&Bs[0][lr][lc] = *(const v16bf*)(gb + k0);
    __syncthreads();
    v16bf bfrg = frag_b(&Bs[0][0][0], 40, wv * 16, 0, lane);
#pragma unroll
    for (int t = 0; t < 8; ++t) {
      v16bf af = frag_a(&As[0][0][0], 40, t * 16, 0, lane);
      acc[t] = wmma_bf16(af, bfrg, acc[t]);
    }
    __syncthreads();
  }
#endif

  // Epilogue: D element (r,lane) -> m = r + 8*(lane>>4), n = lane&15.
  const int nglob = n0 + wv * 16 + (lane & 15);
  const int h = nglob >> 6, d = nglob & 63;
  const float bv = bias[nglob];
#pragma unroll
  for (int t = 0; t < 8; ++t) {
    int mbase = m0 + t * 16 + 8 * (lane >> 4);
#pragma unroll
    for (int r = 0; r < 8; ++r) {
      int m = mbase + r;                 // global row in [0,4096)
      int b = m >> 10, i = m & 1023;
      out[(((size_t)(b * HEADS + h)) * 1024 + i) * DK + d] =
          (bf16)((acc[t][r] + bv) * scale);
    }
  }
}

// ---------------------------------------------------------------------------
// Kernel 5: per-(b,h) scores S = Q K^T + t5_bias, 128x128 tiles.
// grid = (8 jtiles, 8 itiles, 64 bh).  K loop = 64 (2 WMMA steps).
// Epilogue: rel = leaf[j]-leaf[i]; bucket via LDS LUT; +r_emb[bucket,h].
// ---------------------------------------------------------------------------
__global__ __launch_bounds__(256) void score_kernel(
    const bf16* __restrict__ q, const bf16* __restrict__ k,
    const int* __restrict__ leaf, const float* __restrict__ remb,
    const unsigned char* __restrict__ btab, float* __restrict__ out) {
  const int bh = blockIdx.z, b = bh >> 4, h = bh & 15;
  const int i0 = blockIdx.y * 128, j0 = blockIdx.x * 128;
  const int tid = threadIdx.x, lane = tid & 31, wv = tid >> 5;

  __shared__ __align__(32) bf16 Qs[128][72];   // 64 + 8 pad
  __shared__ __align__(32) bf16 Ks[128][72];
  __shared__ int li[128], lj[128];
  __shared__ float rcol[33];
  __shared__ __align__(16) unsigned char tab[4096];

  {  // stage Q/K tiles: 2 threads per row, 32 halfs (64B) each
    int r = tid >> 1, c = (tid & 1) * 32;
    const bf16* sq = q + ((size_t)bh * 1024 + i0 + r) * DK + c;
    const bf16* sk = k + ((size_t)bh * 1024 + j0 + r) * DK + c;
#if HAVE_ASYNC_LDS
    async_copy64(sq, &Qs[r][c]);
    async_copy64(sk, &Ks[r][c]);
#else
    *(v16bf*)&Qs[r][c] = *(const v16bf*)sq;
    *(v16bf*)&Qs[r][c + 16] = *(const v16bf*)(sq + 16);
    *(v16bf*)&Ks[r][c] = *(const v16bf*)sk;
    *(v16bf*)&Ks[r][c + 16] = *(const v16bf*)(sk + 16);
#endif
  }
  if (tid < 128) li[tid] = leaf[b * 1024 + i0 + tid];
  else           lj[tid - 128] = leaf[b * 1024 + j0 + (tid - 128)];
  if (tid < 33) rcol[tid] = remb[tid * HEADS + h];
  ((uint4*)tab)[tid] = ((const uint4*)btab)[tid];    // 256 * 16B = 4KB
#if HAVE_ASYNC_LDS
  wait_async0();
#endif
  __syncthreads();

  v8f acc[8];
#pragma unroll
  for (int t = 0; t < 8; ++t) acc[t] = (v8f)0.0f;

#pragma unroll
  for (int kk = 0; kk < DK; kk += 32) {
    v16bf bfrg = frag_b(&Ks[0][0], 72, wv * 16, kk, lane);
#pragma unroll
    for (int t = 0; t < 8; ++t) {
      v16bf af = frag_a(&Qs[0][0], 72, t * 16, kk, lane);
      acc[t] = wmma_bf16(af, bfrg, acc[t]);
    }
  }

  const int jloc = wv * 16 + (lane & 15);
  const int ljv = lj[jloc];
  const size_t obase = (size_t)bh * 1024 * 1024 + (size_t)(j0 + jloc);
#pragma unroll
  for (int t = 0; t < 8; ++t) {
    int ibase = t * 16 + 8 * (lane >> 4);
#pragma unroll
    for (int r = 0; r < 8; ++r) {
      int iloc = ibase + r;
      int liv = li[iloc];
      int rel = ljv - liv;                           // leaf[j] - leaf[i]
      int bucket = (liv == 0 || ljv == 0) ? NBUCK : (int)tab[rel + 2047];
      out[obase + (size_t)(i0 + iloc) * 1024] = acc[t][r] + rcol[bucket];
    }
  }
}

// ---------------------------------------------------------------------------
extern "C" void kernel_launch(void* const* d_in, const int* in_sizes, int n_in,
                              void* d_out, int out_size, void* d_ws,
                              size_t ws_size, hipStream_t stream) {
  const int*   leaf = (const int*)d_in[0];
  const float* pemb = (const float*)d_in[1];
  const float* remb = (const float*)d_in[2];
  const float* ln_g = (const float*)d_in[3];
  const float* ln_b = (const float*)d_in[4];
  const float* Wq   = (const float*)d_in[5];
  const float* bq   = (const float*)d_in[6];
  const float* Wk   = (const float*)d_in[7];
  const float* bk   = (const float*)d_in[8];
  float* out = (float*)d_out;

  char* w = (char*)d_ws;
  bf16* xn  = (bf16*)(w);                       //  8 MB  [4096][1024]
  bf16* wqb = (bf16*)(w + (8u << 20));          //  2 MB
  bf16* wkb = (bf16*)(w + (10u << 20));         //  2 MB
  bf16* qb  = (bf16*)(w + (12u << 20));         //  8 MB  [bh][1024][64]
  bf16* kb  = (bf16*)(w + (20u << 20));         //  8 MB
  unsigned char* tab = (unsigned char*)(w + (28u << 20));  // 4 KB

  bucket_lut_kernel<<<16, 256, 0, stream>>>(tab);
  cvt_w_kernel<<<4096, 256, 0, stream>>>(Wq, Wk, wqb, wkb);
  ln_kernel<<<512, 256, 0, stream>>>(leaf, pemb, ln_g, ln_b, xn);
  proj_kernel<<<dim3(8, 32, 2), 256, 0, stream>>>(xn, wqb, wkb, bq, bk, qb, kb);
  score_kernel<<<dim3(8, 8, 64), 256, 0, stream>>>(qb, kb, leaf, remb, tab, out);
}
